// MultiHeadAttention_67637144977483
// MI455X (gfx1250) — matrix operand order repaired and checked by reading, not compiled, not run on hardware
//
#include <hip/hip_runtime.h>
#include <stdint.h>

// ---------------------------------------------------------------------------
// Types / helpers
// ---------------------------------------------------------------------------
typedef __attribute__((ext_vector_type(16))) __bf16 v16bf;
typedef __attribute__((ext_vector_type(8)))  float  v8f;

struct __align__(16) u128 { uint32_t x, y, z, w; };

__device__ __forceinline__ __bf16 f2bf(float f) {
  uint32_t u = __builtin_bit_cast(uint32_t, f);
  u += 0x7FFFu + ((u >> 16) & 1u);               // round-to-nearest-even
  uint16_t h = (uint16_t)(u >> 16);
  return __builtin_bit_cast(__bf16, h);
}

__global__ __launch_bounds__(256)
void cvt_f32_bf16(const float* __restrict__ src, __bf16* __restrict__ dst, int n) {
  int i = blockIdx.x * 256 + threadIdx.x;
  if (i < n) dst[i] = f2bf(src[i]);
}

// ---------------------------------------------------------------------------
// Tensor Data Mover tile load (gfx1250): 2-D tile of bf16 rows -> LDS with
// row padding (16 DWORDs data + 4 DWORDs pad == 80B LDS row pitch).
// D# layout per CDNA5 ISA.
// This toolchain exposes the 6-arg builtin:
//   (uint32x4 g0, int32x8 g1, int32x4 g2, int32x4 g3, int32x8 extra, i32 cpol)
// ---------------------------------------------------------------------------
#if defined(__gfx1250__) && defined(__has_builtin)
#if __has_builtin(__builtin_amdgcn_tensor_load_to_lds) && \
    __has_builtin(__builtin_amdgcn_s_wait_tensorcnt)
#define TDM_OK 1
#endif
#endif

#ifdef TDM_OK
typedef unsigned int u32x4 __attribute__((ext_vector_type(4)));
typedef int          i32x8 __attribute__((ext_vector_type(8)));
typedef int          i32x4 __attribute__((ext_vector_type(4)));

__device__ __forceinline__ void tdm_load_tile_2d(
    const __bf16* gsrc,   // global address of tile start
    uint32_t lds_addr,    // LDS byte offset of tile start
    uint32_t rows_rem,    // tensor_dim1 (remaining rows; OOB rows read as 0)
    uint32_t cols_rem,    // tensor_dim0 (remaining cols)
    uint32_t row_pitch)   // elements between consecutive rows
{
  uint64_t ga = (uint64_t)(uintptr_t)gsrc;
  u32x4 g0;
  g0[0] = 1u;                                            // count=1 (valid), user mode
  g0[1] = lds_addr;                                      // lds_addr [63:32]
  g0[2] = (uint32_t)ga;                                  // global_addr lo
  g0[3] = (uint32_t)((ga >> 32) & 0x01FFFFFFu)           // global_addr [56:32]
        | (2u << 30);                                    // type=2 ("image")
  i32x8 g1;
  g1[0] = (int)((1u << 16)        // data_size = 1 -> 2 bytes (bf16)
              | (1u << 20)        // pad_enable
              | (3u << 22)        // pad_interval: 16 DWORDs between pads
              | (3u << 25));      // pad_amount:   4 DWORDs of pad
  g1[1] = (int)((cols_rem & 0xFFFFu) << 16);                               // tensor_dim0 lo16
  g1[2] = (int)(((cols_rem >> 16) & 0xFFFFu) | ((rows_rem & 0xFFFFu) << 16)); // dim0 hi | dim1 lo
  g1[3] = (int)(((rows_rem >> 16) & 0xFFFFu) | (32u << 16));               // dim1 hi | tile_dim0=32
  g1[4] = (int)128u;                                                       // tile_dim1=128, tile_dim2=0
  g1[5] = (int)row_pitch;                                                  // tensor_dim0_stride lo32
  g1[6] = 0;                                                               // stride hi, dim1_stride lo
  g1[7] = 0;
  i32x4 z4 = {0, 0, 0, 0};
  i32x8 z8 = {0, 0, 0, 0, 0, 0, 0, 0};
  __builtin_amdgcn_tensor_load_to_lds(g0, g1, z4, z4, z8, 0);
}
#endif

// ---------------------------------------------------------------------------
// Tiled bf16 WMMA GEMM:  C[M,N] = act( alpha * A[M,K] x B + bias[M] )
//   A: row-major [M,K] bf16.
//   B: TRANSB=0 -> row-major [K,N]; TRANSB=1 -> row-major [N,K] (i.e. C=A*B^T)
//   Block tile 128x128, K-step 32, 8 waves each computing a 64x32 subtile
//   (4x2 grid of 16x16 WMMA accumulators).
// ---------------------------------------------------------------------------
#define BM   128
#define BNT  128
#define BKK  32
#define LDSP 40   // BKK + 8 bf16 pad (80B rows, still 16B aligned)

template<bool TRANSB, bool HAS_BIAS, bool LEAKY, bool OUT_BF16, bool MGUARD>
__global__ __launch_bounds__(256)
void gemm_wmma_bf16(const __bf16* __restrict__ A, const __bf16* __restrict__ Bm,
                    const float* __restrict__ bias, void* __restrict__ Cout,
                    int M, int N, int K,
                    long sA, long sB, long sC, float alpha) {
  __shared__ __align__(16) __bf16 As[BM][LDSP];
  __shared__ __align__(16) __bf16 Bs[BNT][LDSP];   // stored transposed: Bs[n][k]

  const int t     = threadIdx.x;
  const int lane  = t & 31;
  const int wave  = t >> 5;
  const int wm    = wave >> 2;       // 0..1 : 64-row block
  const int wn    = wave & 3;        // 0..3 : 32-col block
  const int l16   = lane & 15;
  const int lhalf = lane >> 4;
  const int koffA = lhalf * 8;       // A lane K pattern: {koffA..+7, 16+koffA..+7}

  const int nb = blockIdx.x * BNT;
  const int mb = blockIdx.y * BM;
  const int z  = blockIdx.z;

  const __bf16* Ab = A  + (size_t)z * sA;
  const __bf16* Bb = Bm + (size_t)z * sB;

#ifdef TDM_OK
  const uint32_t ldsA = (uint32_t)(uintptr_t)(&As[0][0]);
  const uint32_t ldsB = (uint32_t)(uintptr_t)(&Bs[0][0]);
#endif

  v8f acc[4][2];
#pragma unroll
  for (int i = 0; i < 4; ++i)
#pragma unroll
    for (int j = 0; j < 2; ++j)
#pragma unroll
      for (int v = 0; v < 8; ++v) acc[i][j][v] = 0.0f;

  for (int kb = 0; kb < K; kb += BKK) {
#ifdef TDM_OK
    // ---- A tile (and B tile when it is a straight copy) via the TDM ----
    if (wave == 0) {
      tdm_load_tile_2d(Ab + (size_t)mb * K + kb, ldsA,
                       (uint32_t)(M - mb), (uint32_t)(K - kb), (uint32_t)K);
      if (TRANSB)
        tdm_load_tile_2d(Bb + (size_t)nb * K + kb, ldsB,
                         (uint32_t)(N - nb), (uint32_t)(K - kb), (uint32_t)K);
    }
#else
    // ---- manual A tile (128 x 32), zero-pad ragged M ----
#pragma unroll
    for (int r = 0; r < 2; ++r) {
      int idx = t + r * 256;               // 0..511
      int row = idx >> 2;
      int c8  = (idx & 3) << 3;
      u128 val = {0u, 0u, 0u, 0u};
      int gr = mb + row;
      if (!MGUARD || gr < M) {
        val = *(const u128*)(Ab + (size_t)gr * K + kb + c8);
        if (kb + BKK < K)
          __builtin_prefetch(Ab + (size_t)gr * K + kb + BKK + c8, 0, 1);
      }
      *(u128*)(&As[row][c8]) = val;
    }
    if (TRANSB) {
      // source is [N,K]: rows are n, contiguous along k
#pragma unroll
      for (int r = 0; r < 2; ++r) {
        int idx = t + r * 256;
        int row = idx >> 2;
        int c8  = (idx & 3) << 3;
        u128 val = *(const u128*)(Bb + (size_t)(nb + row) * K + kb + c8);
        *(u128*)(&Bs[row][c8]) = val;
        if (kb + BKK < K)
          __builtin_prefetch(Bb + (size_t)(nb + row) * K + kb + BKK + c8, 0, 1);
      }
    }
#endif
    if (!TRANSB) {
      // source is [K,N]: vector-load along n, scatter-transpose into LDS
#pragma unroll
      for (int r = 0; r < 2; ++r) {
        int idx = t + r * 256;
        int kr  = idx >> 4;                // 0..31
        int n8  = (idx & 15) << 3;         // 0..120
        union { u128 q; __bf16 h[8]; } val;
        val.q = *(const u128*)(Bb + (size_t)(kb + kr) * N + nb + n8);
#pragma unroll
        for (int e = 0; e < 8; ++e) Bs[n8 + e][kr] = val.h[e];
        if (kb + BKK < K)
          __builtin_prefetch(Bb + (size_t)(kb + BKK + kr) * N + nb + n8, 0, 1);
      }
    }
#ifdef TDM_OK
    if (wave == 0) __builtin_amdgcn_s_wait_tensorcnt((short)0);
#endif
    __syncthreads();

    // ---- fragments ----
    union Frag { v16bf v; u128 q[2]; };
    Frag fa[4], fb[2];
#pragma unroll
    for (int i = 0; i < 4; ++i) {
      // A 16x32 layout: lanes 0-15: K 0..7 & 16..23 ; lanes 16-31: K 8..15 & 24..31
      const __bf16* p = &As[wm * 64 + i * 16 + l16][0];
      fa[i].q[0] = *(const u128*)(p + koffA);
      fa[i].q[1] = *(const u128*)(p + 16 + koffA);
    }
#pragma unroll
    for (int j = 0; j < 2; ++j) {
      // B 32x16 layout: lanes 0-15: K 0..7 & 16..23 ; lanes 16-31: K 8..15 & 24..31
      const __bf16* p = &Bs[wn * 32 + j * 16 + l16][0];
      fb[j].q[0] = *(const u128*)(p + koffA);
      fb[j].q[1] = *(const u128*)(p + 16 + koffA);
    }
#pragma unroll
    for (int i = 0; i < 4; ++i)
#pragma unroll
      for (int j = 0; j < 2; ++j)
        acc[i][j] = __builtin_amdgcn_wmma_f32_16x16x32_bf16(
            false, fa[i].v, false, fb[j].v, (short)0, acc[i][j], false, false);
    __syncthreads();
  }

  // ---- epilogue: D layout -> row = v + 8*lhalf, col = l16 within 16x16 tile ----
  float*  Cf = (float*)Cout  + (size_t)z * sC;
  __bf16* Cb = (__bf16*)Cout + (size_t)z * sC;
#pragma unroll
  for (int i = 0; i < 4; ++i) {
#pragma unroll
    for (int j = 0; j < 2; ++j) {
      int row0 = mb + wm * 64 + i * 16 + 8 * lhalf;
      int col  = nb + wn * 32 + j * 16 + l16;
#pragma unroll
      for (int v = 0; v < 8; ++v) {
        int row = row0 + v;
        if (MGUARD && row >= M) continue;
        float x = acc[i][j][v] * alpha;
        if (HAS_BIAS) x += bias[row];
        if (LEAKY) x = (x >= 0.0f) ? x : 0.01f * x;
        if (OUT_BF16) Cb[(size_t)row * N + col] = f2bf(x);
        else          Cf[(size_t)row * N + col] = x;
      }
    }
  }
}

// ---------------------------------------------------------------------------
// Softmax over rows of length 512 (one block per row), bf16 output
// ---------------------------------------------------------------------------
__global__ __launch_bounds__(256)
void softmax_rows512(const float* __restrict__ s, __bf16* __restrict__ attn) {
  __shared__ float red[256];
  const int row = blockIdx.x;
  const float* src = s + (size_t)row * 512;
  const int t = threadIdx.x;
  float a0 = src[t], a1 = src[t + 256];
  red[t] = fmaxf(a0, a1);
  __syncthreads();
  for (int off = 128; off > 0; off >>= 1) {
    if (t < off) red[t] = fmaxf(red[t], red[t + off]);
    __syncthreads();
  }
  float mx = red[0];
  __syncthreads();
  float e0 = __expf(a0 - mx), e1 = __expf(a1 - mx);
  red[t] = e0 + e1;
  __syncthreads();
  for (int off = 128; off > 0; off >>= 1) {
    if (t < off) red[t] += red[t + off];
    __syncthreads();
  }
  float inv = 1.0f / red[0];
  __bf16* dst = attn + (size_t)row * 512;
  dst[t]       = f2bf(e0 * inv);
  dst[t + 256] = f2bf(e1 * inv);
}

// ---------------------------------------------------------------------------
// BatchNorm stats: channel c2 of the permuted tensor gathers columns
// [8*c2, 8*c2+8) of out[b, 0..511, :] over all 8 batches (32768 samples).
// ---------------------------------------------------------------------------
__global__ __launch_bounds__(256)
void bn_stats(const float* __restrict__ out, float* __restrict__ stat) {
  __shared__ float r1[256], r2[256];
  const int c2 = blockIdx.x;
  const int t = threadIdx.x;
  float s1 = 0.0f, s2 = 0.0f;
  for (int idx = t; idx < 32768; idx += 256) {
    int b   = idx >> 12;
    int rem = idx & 4095;
    int c   = rem >> 3;
    int ml  = rem & 7;
    float v = out[(((size_t)b * 512 + c) << 12) + (c2 << 3) + ml];
    s1 += v; s2 += v * v;
  }
  r1[t] = s1; r2[t] = s2;
  __syncthreads();
  for (int off = 128; off > 0; off >>= 1) {
    if (t < off) { r1[t] += r1[t + off]; r2[t] += r2[t + off]; }
    __syncthreads();
  }
  if (t == 0) {
    float mean = r1[0] * (1.0f / 32768.0f);
    float var  = r2[0] * (1.0f / 32768.0f) - mean * mean;
    stat[c2 * 2]     = mean;
    stat[c2 * 2 + 1] = rsqrtf(var + 1e-4f);
  }
}

// ---------------------------------------------------------------------------
// Normalize + scale/shift + leaky + permute (out[b,c,m] -> y[b,c2,j]) -> bf16
//   c2 = m>>3, j = ((m&7)<<9)|c
// ---------------------------------------------------------------------------
__global__ __launch_bounds__(256)
void bn_apply_perm(const float* __restrict__ out, const float* __restrict__ stat,
                   const float* __restrict__ g, const float* __restrict__ beta,
                   __bf16* __restrict__ outn) {
  const int i = blockIdx.x * 256 + threadIdx.x;   // 0 .. 16777215
  const int b  = i >> 21;
  const int r  = i & ((1 << 21) - 1);
  const int c  = r >> 12;
  const int m  = r & 4095;
  const int c2 = m >> 3;
  const int j  = ((m & 7) << 9) | c;
  float x = out[i];
  float y = (x - stat[c2 * 2]) * stat[c2 * 2 + 1] * g[c2] + beta[c2];
  y = (y >= 0.0f) ? y : 0.01f * y;
  outn[((((size_t)b << 9) | (size_t)c2) << 12) + j] = f2bf(y);
}

// ---------------------------------------------------------------------------
// Host orchestration
// ---------------------------------------------------------------------------
extern "C" void kernel_launch(void* const* d_in, const int* in_sizes, int n_in,
                              void* d_out, int out_size, void* d_ws, size_t ws_size,
                              hipStream_t stream) {
  (void)in_sizes; (void)n_in; (void)out_size;
  const float* q    = (const float*)d_in[0];
  const float* k    = (const float*)d_in[1];
  const float* v    = (const float*)d_in[2];
  const float* wq   = (const float*)d_in[3];
  const float* bq   = (const float*)d_in[4];
  const float* wk   = (const float*)d_in[5];
  const float* bk   = (const float*)d_in[6];
  const float* wv   = (const float*)d_in[7];
  const float* bv   = (const float*)d_in[8];
  const float* bn_g = (const float*)d_in[9];
  const float* bn_b = (const float*)d_in[10];
  const float* w1   = (const float*)d_in[11];
  const float* b1   = (const float*)d_in[12];
  const float* w2   = (const float*)d_in[13];
  const float* b2   = (const float*)d_in[14];
  float* outp = (float*)d_out;

  // Sizes: B=8, C=64, CN=512, HW=4096
  const size_t SZ_P  = (size_t)8 * 512 * 4096 * 2;  // 33,554,432  bf16 [B,CN,HW]
  const size_t SZ_X  = (size_t)8 * 64  * 4096 * 2;  //  4,194,304  bf16 [B,C,HW]
  const size_t SZ_S  = (size_t)8 * 512 * 512 * 4;   //  8,388,608  f32  scores
  const size_t SZ_O  = (size_t)8 * 512 * 4096 * 4;  // 67,108,864  f32  attn*V
  const size_t SZ_W  = (size_t)512 * 64 * 2;        //     65,536
  const size_t SZ_W1 = (size_t)512 * 512 * 2;       //    524,288

  char* ws = (char*)d_ws;
  const size_t NEED = 3 * SZ_P + SZ_P + 3 * SZ_W + SZ_W1 + SZ_W + 4096;
  if (ws_size < NEED) return;  // insufficient scratch; cannot allocate in capture

  // Region A [0 .. 96M): qp | kp | vp ; later reused as outbuf(64M) | outn(32M)
  __bf16* qp     = (__bf16*)(ws);
  __bf16* kp     = (__bf16*)(ws + SZ_P);
  __bf16* vp     = (__bf16*)(ws + 2 * SZ_P);
  float*  outbuf = (float*)(ws);                 // overlays qp+kp (dead by then)
  __bf16* outn   = (__bf16*)(ws + SZ_O);         // overlays vp   (dead by then)
  // Region B [96M .. 128M): xq|xk|xv|scores|attn ; later reused as h1
  char* rb = ws + 3 * SZ_P;
  __bf16* xq     = (__bf16*)(rb);
  __bf16* xk     = (__bf16*)(rb + SZ_X);
  __bf16* xv     = (__bf16*)(rb + 2 * SZ_X);
  float*  scores = (float*)(rb + 3 * SZ_X);
  __bf16* attn   = (__bf16*)(rb + 3 * SZ_X + SZ_S);
  __bf16* h1     = (__bf16*)(rb);                // overlays xq..attn (dead by then)
  // Region C [128M ..): weights + bn stats (live whole launch)
  char* rc = rb + SZ_P;
  __bf16* wqb = (__bf16*)(rc);
  __bf16* wkb = (__bf16*)(rc + SZ_W);
  __bf16* wvb = (__bf16*)(rc + 2 * SZ_W);
  __bf16* w1b = (__bf16*)(rc + 3 * SZ_W);
  __bf16* w2b = (__bf16*)(rc + 3 * SZ_W + SZ_W1);
  float* bnstat = (float*)(rc + 3 * SZ_W + SZ_W1 + SZ_W);

  const dim3 blk(256);
  const int NX = 8 * 64 * 4096;       // 2,097,152
  const int NW = 512 * 64;            // 32,768
  const int NW1 = 512 * 512;          // 262,144

  // 1) fp32 -> bf16 conversions
  cvt_f32_bf16<<<dim3(NX / 256), blk, 0, stream>>>(q, xq, NX);
  cvt_f32_bf16<<<dim3(NX / 256), blk, 0, stream>>>(k, xk, NX);
  cvt_f32_bf16<<<dim3(NX / 256), blk, 0, stream>>>(v, xv, NX);
  cvt_f32_bf16<<<dim3(NW / 256), blk, 0, stream>>>(wq, wqb, NW);
  cvt_f32_bf16<<<dim3(NW / 256), blk, 0, stream>>>(wk, wkb, NW);
  cvt_f32_bf16<<<dim3(NW / 256), blk, 0, stream>>>(wv, wvb, NW);
  cvt_f32_bf16<<<dim3(NW1 / 256), blk, 0, stream>>>(w1, w1b, NW1);
  cvt_f32_bf16<<<dim3(NW / 256), blk, 0, stream>>>(w2, w2b, NW);

  // 2) projections: [512,64] x [64,4096] + bias -> bf16   (per batch)
  const dim3 gproj(4096 / BNT, 512 / BM, 8);
  gemm_wmma_bf16<false, true, false, true, false><<<gproj, blk, 0, stream>>>(
      wqb, xq, bq, qp, 512, 4096, 64, 0, (long)64 * 4096, (long)512 * 4096, 1.0f);
  gemm_wmma_bf16<false, true, false, true, false><<<gproj, blk, 0, stream>>>(
      wkb, xk, bk, kp, 512, 4096, 64, 0, (long)64 * 4096, (long)512 * 4096, 1.0f);
  gemm_wmma_bf16<false, true, false, true, false><<<gproj, blk, 0, stream>>>(
      wvb, xv, bv, vp, 512, 4096, 64, 0, (long)64 * 4096, (long)512 * 4096, 1.0f);

  // 3) scores = (Q x K^T) / sqrt(HW)  : [512,4096] x [4096,512] -> f32
  const dim3 gsc(512 / BNT, 512 / BM, 8);
  gemm_wmma_bf16<true, false, false, false, false><<<gsc, blk, 0, stream>>>(
      qp, kp, nullptr, scores, 512, 512, 4096,
      (long)512 * 4096, (long)512 * 4096, (long)512 * 512, 1.0f / 64.0f);

  // 4) softmax over channel dim (rows of 512)
  softmax_rows512<<<dim3(8 * 512), blk, 0, stream>>>(scores, attn);

  // 5) out = attn x V : [512,512] x [512,4096] -> f32 (into reused region A)
  gemm_wmma_bf16<false, false, false, false, false><<<gproj, blk, 0, stream>>>(
      attn, vp, nullptr, outbuf, 512, 4096, 512,
      (long)512 * 512, (long)512 * 4096, (long)512 * 4096, 1.0f);

  // 6) BatchNorm (training-mode batch stats) fused with the permute + leaky
  bn_stats<<<dim3(512), blk, 0, stream>>>(outbuf, bnstat);
  bn_apply_perm<<<dim3((8 * 512 * 4096) / 256), blk, 0, stream>>>(
      outbuf, bnstat, bn_g, bn_b, outn);

  // 7) h1 = leaky(W1 x y + b1) : [512,512] x [512,4096] -> bf16
  gemm_wmma_bf16<false, true, true, true, false><<<gproj, blk, 0, stream>>>(
      w1b, outn, b1, h1, 512, 4096, 512, 0, (long)512 * 4096, (long)512 * 4096, 1.0f);

  // 8) out = W2 x h1 + b2 : [64,512] x [512,4096] -> f32 d_out  (ragged M)
  const dim3 gw2(4096 / BNT, 1, 8);
  gemm_wmma_bf16<false, true, false, false, true><<<gw2, blk, 0, stream>>>(
      w2b, h1, b2, outp, 64, 4096, 512, 0, (long)512 * 4096, (long)64 * 4096, 1.0f);
}
